// NodeSelfAttention_8615704395977
// MI455X (gfx1250) — matrix operand order repaired and checked by reading, not compiled, not run on hardware
//
#include <hip/hip_runtime.h>
#include <hip/hip_bf16.h>
#include <math.h>

typedef __bf16 bf16_t;
typedef __attribute__((ext_vector_type(16))) __bf16 v16bf;
typedef __attribute__((ext_vector_type(8)))  __bf16 v8bf;
typedef __attribute__((ext_vector_type(8)))  float   v8f;

// ---------------------------------------------------------------------------
// WMMA wrapper: D(16x16,f32) = A(16x32,bf16) * B(32x16,bf16) + C
// ---------------------------------------------------------------------------
__device__ __forceinline__ v8f wmma_bf16(v16bf a, v16bf b, v8f c) {
  return __builtin_amdgcn_wmma_f32_16x16x32_bf16(false, a, false, b, (short)0, c,
                                                 false, false);
}

__device__ __forceinline__ float gelu_exact(float x) {
  return 0.5f * x * (1.0f + erff(x * 0.70710678118654752440f));
}

__device__ __forceinline__ v16bf cat8(v8bf lo, v8bf hi) {
  return __builtin_shufflevector(lo, hi, 0, 1, 2, 3, 4, 5, 6, 7,
                                 8, 9, 10, 11, 12, 13, 14, 15);
}

// ---------------------------------------------------------------------------
// Fragment loaders (wave32, 16x16x32 bf16 layouts per CDNA5 ISA 7.12.2)
// A (16M x 32K) row-major: lane m=l&15; elems j<8 -> k=(l>>4)*8+j ; j>=8 -> k=16+(l>>4)*8+(j-8)
// Two contiguous 16-byte chunks per lane.
// ---------------------------------------------------------------------------
__device__ __forceinline__ v16bf load_a_frag(const bf16_t* __restrict__ A, int lda,
                                             int row0, int k0, int lane) {
  int m  = row0 + (lane & 15);
  int hb = (lane >> 4) * 8;
  const bf16_t* p = A + (size_t)m * lda + k0 + hb;
  return cat8(*(const v8bf*)(p), *(const v8bf*)(p + 16));
}

// B fragment from a matrix stored transposed, Bt[N][K] row-major:
// lane n = n0 + (l&15) (a row of Bt); elems j<8 -> k = k0 + (l>>4)*8 + j ; j>=8 -> k = k0 + 16 + (l>>4)*8 + (j-8).
// Two contiguous 16-byte chunks per lane (same k order as A).
__device__ __forceinline__ v16bf load_bT_frag(const bf16_t* __restrict__ Bt, int ldt,
                                              int n0, int k0, int lane) {
  const bf16_t* p = Bt + (size_t)(n0 + (lane & 15)) * ldt + k0 + (lane >> 4) * 8;
  return cat8(*(const v8bf*)(p), *(const v8bf*)(p + 16));
}

// ---------------------------------------------------------------------------
// Utility kernels
// ---------------------------------------------------------------------------
__global__ void zero_f32_kernel(float* __restrict__ p, long n) {
  long i = (long)blockIdx.x * blockDim.x + threadIdx.x;
  long stride = (long)gridDim.x * blockDim.x;
  for (; i < n; i += stride) p[i] = 0.0f;
}

// fp32 [K,N] -> bf16 transposed [N,K], 32x32 LDS tiles for coalescing.
__global__ __launch_bounds__(256) void cvt_transpose_kernel(const float* __restrict__ in,
                                                            bf16_t* __restrict__ out,
                                                            int K, int N) {
  __shared__ float tile[32][33];
  int kb = blockIdx.y * 32, nb = blockIdx.x * 32;
  int tx = threadIdx.x & 31, ty = threadIdx.x >> 5;  // 32 x 8
#pragma unroll
  for (int i = ty; i < 32; i += 8)
    tile[i][tx] = in[(size_t)(kb + i) * N + nb + tx];
  __syncthreads();
#pragma unroll
  for (int i = ty; i < 32; i += 8)
    out[(size_t)(nb + i) * K + kb + tx] = (bf16_t)tile[tx][i];
}

// LayerNorm: one 256-thread block per row, bf16 output.
__global__ __launch_bounds__(256) void ln_kernel(const float* __restrict__ x,
                                                 const float* __restrict__ g,
                                                 const float* __restrict__ bb,
                                                 bf16_t* __restrict__ out, int Dm) {
  __shared__ float sred[16];
  size_t row = blockIdx.x;
  const float* xr = x + row * Dm;
  float s = 0.0f, ss = 0.0f;
  for (int i = threadIdx.x; i < Dm; i += 256) { float v = xr[i]; s += v; ss += v * v; }
#pragma unroll
  for (int m = 16; m >= 1; m >>= 1) { s += __shfl_xor(s, m, 32); ss += __shfl_xor(ss, m, 32); }
  int wave = threadIdx.x >> 5, lane = threadIdx.x & 31;
  if (lane == 0) { sred[wave] = s; sred[8 + wave] = ss; }
  __syncthreads();
  float ts = 0.0f, tss = 0.0f;
#pragma unroll
  for (int wv = 0; wv < 8; ++wv) { ts += sred[wv]; tss += sred[8 + wv]; }
  float mean = ts / Dm;
  float var  = tss / Dm - mean * mean;
  float inv  = rsqrtf(var + 1e-5f);
  for (int i = threadIdx.x; i < Dm; i += 256)
    out[row * Dm + i] = (bf16_t)((xr[i] - mean) * inv * g[i] + bb[i]);
}

// Edge MLP: proj[e,h] = W2[h] . gelu(edge_attr[e]*W1[h] + b1[h]) + b2[h]
// scattered into bias[h, src[e], dst[e]]
__global__ void edge_bias_kernel(const float* __restrict__ edge_attr,
                                 const int* __restrict__ edge_index,
                                 const float* __restrict__ W1, const float* __restrict__ b1,
                                 const float* __restrict__ W2, const float* __restrict__ b2,
                                 float* __restrict__ bias, int E, int H, int S) {
  int idx = blockIdx.x * blockDim.x + threadIdx.x;
  if (idx >= E * H) return;
  int e = idx / H, h = idx % H;
  float ea  = edge_attr[e];
  float acc = b2[h];
#pragma unroll
  for (int k = 0; k < 16; ++k) {
    float t = ea * W1[h * 16 + k] + b1[h * 16 + k];
    acc += gelu_exact(t) * W2[h * 16 + k];
  }
  int src = edge_index[e];
  int dst = edge_index[E + e];
  bias[(size_t)h * S * S + (size_t)src * S + dst] = acc;
}

// ---------------------------------------------------------------------------
// WMMA-bf16 GEMM with transposed weights: C[M,N] = act(A[M,K] @ Bt^T + bias) + residual
// Bt is [N,K] row-major. Block = 256 threads (8 waves), block tile 256x128,
// wave tile 64x64 (16 WMMAs / 16 b128 loads per K-step for 2x operand reuse).
// act: 0 = none, 1 = exact GELU. Optional outputs: f32 [M,N], bf16 [M,N], bf16^T [N,M].
// ---------------------------------------------------------------------------
__global__ __launch_bounds__(256) void gemm_bf16_kernel(
    const bf16_t* __restrict__ A, const bf16_t* __restrict__ Bt,
    const float* __restrict__ bias, const float* __restrict__ residual,
    float* __restrict__ outf, bf16_t* __restrict__ outb, bf16_t* __restrict__ outbT,
    int M, int N, int K, int act) {
  int lane = threadIdx.x & 31;
  int wave = threadIdx.x >> 5;    // 0..7
  int wm = wave & 3;              // 0..3 -> 64 rows each
  int wn = wave >> 2;             // 0..1 -> 64 cols each
  int row0 = blockIdx.y * 256 + wm * 64;
  int col0 = blockIdx.x * 128 + wn * 64;

  v8f acc[4][4];
#pragma unroll
  for (int mi = 0; mi < 4; ++mi)
#pragma unroll
    for (int t = 0; t < 4; ++t) acc[mi][t] = (v8f)0.0f;

  for (int k0 = 0; k0 < K; k0 += 32) {
    if (k0 + 32 < K) {
      __builtin_prefetch(A + (size_t)(row0 + (lane & 15)) * K + k0 + 32, 0, 3);
      __builtin_prefetch(Bt + (size_t)(col0 + (lane & 15)) * K + k0 + 32, 0, 3);
    }
    v16bf af[4];
#pragma unroll
    for (int mi = 0; mi < 4; ++mi)
      af[mi] = load_a_frag(A, K, row0 + mi * 16, k0, lane);
    v16bf bfr[4];
#pragma unroll
    for (int t = 0; t < 4; ++t)
      bfr[t] = load_bT_frag(Bt, K, col0 + t * 16, k0, lane);
#pragma unroll
    for (int mi = 0; mi < 4; ++mi)
#pragma unroll
      for (int t = 0; t < 4; ++t)
        acc[mi][t] = wmma_bf16(af[mi], bfr[t], acc[mi][t]);
  }

  int hi = lane >> 4, nl = lane & 15;
#pragma unroll
  for (int mi = 0; mi < 4; ++mi)
#pragma unroll
    for (int t = 0; t < 4; ++t)
#pragma unroll
      for (int r = 0; r < 8; ++r) {
        int row = row0 + mi * 16 + r + hi * 8;   // C layout: vgpr r -> M=r / r+8
        int col = col0 + t * 16 + nl;
        float v = acc[mi][t][r];
        if (bias)     v += bias[col];
        if (act == 1) v  = gelu_exact(v);
        if (residual) v += residual[(size_t)row * N + col];
        if (outf)  outf[(size_t)row * N + col] = v;
        if (outb)  outb[(size_t)row * N + col] = (bf16_t)v;
        if (outbT) outbT[(size_t)col * M + row] = (bf16_t)v;
      }
}

// ---------------------------------------------------------------------------
// Flash attention with mask + edge bias.
// grid = (S/64, H, B), block = 128 (4 waves). Each wave owns 16 query rows.
// q/k stored [B,S,D] bf16 (head h at cols [h*hd,(h+1)*hd)); v stored transposed [D, B*S].
// ---------------------------------------------------------------------------
__global__ __launch_bounds__(128) void attn_kernel(
    const bf16_t* __restrict__ qbuf, const bf16_t* __restrict__ kbuf,
    const bf16_t* __restrict__ vT, const unsigned char* __restrict__ mask,
    const float* __restrict__ biasHSS, bf16_t* __restrict__ outb,
    int S, int Dm, int hd, int Mtot) {
  int qblk = blockIdx.x, h = blockIdx.y, b = blockIdx.z;
  int wave = threadIdx.x >> 5, lane = threadIdx.x & 31;
  int hi = lane >> 4, nl = lane & 15;
  int qbase = qblk * 64 + wave * 16;

  __shared__ __align__(16) bf16_t pshare[4][16][64];

  const bf16_t* Qb  = qbuf + (size_t)b * S * Dm + (size_t)h * hd;
  const bf16_t* Kb  = kbuf + (size_t)b * S * Dm + (size_t)h * hd;
  const bf16_t* Vtb = vT + (size_t)h * hd * Mtot + (size_t)b * S;  // [d][key], stride Mtot

  // Q fragments, loaded once (hd = 64 = 2 K-steps of 32)
  v16bf qf[2];
  qf[0] = load_a_frag(Qb, Dm, qbase, 0, lane);
  qf[1] = load_a_frag(Qb, Dm, qbase, 32, lane);

  v8f oacc[4];
#pragma unroll
  for (int t = 0; t < 4; ++t) oacc[t] = (v8f)0.0f;
  float mrun[8], lrun[8];
#pragma unroll
  for (int r = 0; r < 8; ++r) { mrun[r] = -INFINITY; lrun[r] = 0.0f; }

  const float scale = 0.125f;  // 1/sqrt(64)

  for (int kb = 0; kb < S; kb += 64) {
    // ---- scores: Q . K^T (4 key-tiles x 2 d-steps); K fragments contiguous ----
    v8f sc[4];
#pragma unroll
    for (int t = 0; t < 4; ++t) sc[t] = (v8f)0.0f;
#pragma unroll
    for (int ds = 0; ds < 2; ++ds) {
#pragma unroll
      for (int t = 0; t < 4; ++t) {
        v16bf kf = load_bT_frag(Kb, Dm, kb + t * 16, ds * 32, lane);
        sc[t] = wmma_bf16(qf[ds], kf, sc[t]);
      }
    }

    // ---- mask + edge bias + scale ----
#pragma unroll
    for (int t = 0; t < 4; ++t)
#pragma unroll
      for (int r = 0; r < 8; ++r) {
        int qg = qbase + r + hi * 8;
        int kg = kb + t * 16 + nl;
        float s = sc[t][r] * scale;
        bool mk = mask[((size_t)b * S + qg) * S + kg] != 0;
        s = mk ? (s + biasHSS[((size_t)h * S + qg) * S + kg]) : -1e9f;
        sc[t][r] = s;
      }

    // ---- online softmax ----
    float mblk[8];
#pragma unroll
    for (int r = 0; r < 8; ++r) {
      float m = sc[0][r];
#pragma unroll
      for (int t = 1; t < 4; ++t) m = fmaxf(m, sc[t][r]);
#pragma unroll
      for (int msk = 8; msk >= 1; msk >>= 1) m = fmaxf(m, __shfl_xor(m, msk, 32));
      mblk[r] = m;
    }
    float sfac[8];
#pragma unroll
    for (int r = 0; r < 8; ++r) {
      float mnew = fmaxf(mrun[r], mblk[r]);
      sfac[r] = __expf(mrun[r] - mnew);
      mrun[r] = mnew;
    }
    float lblk[8];
#pragma unroll
    for (int r = 0; r < 8; ++r) lblk[r] = 0.0f;
#pragma unroll
    for (int t = 0; t < 4; ++t)
#pragma unroll
      for (int r = 0; r < 8; ++r) {
        float p = __expf(sc[t][r] - mrun[r]);
        sc[t][r] = p;
        lblk[r] += p;
      }
#pragma unroll
    for (int r = 0; r < 8; ++r) {
      float l = lblk[r];
#pragma unroll
      for (int msk = 8; msk >= 1; msk >>= 1) l += __shfl_xor(l, msk, 32);
      lrun[r] = lrun[r] * sfac[r] + l;
    }
#pragma unroll
    for (int t = 0; t < 4; ++t)
#pragma unroll
      for (int r = 0; r < 8; ++r) oacc[t][r] *= sfac[r];

    // ---- stage P through LDS: C-layout -> A-layout ----
    __syncthreads();
#pragma unroll
    for (int t = 0; t < 4; ++t)
#pragma unroll
      for (int r = 0; r < 8; ++r)
        pshare[wave][r + hi * 8][t * 16 + nl] = (bf16_t)sc[t][r];
    __syncthreads();

    // ---- O += P @ V (V fragments contiguous via V^T) ----
#pragma unroll
    for (int ks = 0; ks < 2; ++ks) {
      const bf16_t* pw = &pshare[wave][lane & 15][ks * 32 + hi * 8];
      v16bf pa = cat8(*(const v8bf*)(pw), *(const v8bf*)(pw + 16));
#pragma unroll
      for (int t = 0; t < 4; ++t) {
        v16bf vf = load_bT_frag(Vtb, Mtot, t * 16, kb + ks * 32, lane);
        oacc[t] = wmma_bf16(pa, vf, oacc[t]);
      }
    }
  }

  // ---- normalize and write [B,S,D] bf16 ----
#pragma unroll
  for (int t = 0; t < 4; ++t)
#pragma unroll
    for (int r = 0; r < 8; ++r) {
      int qg = qbase + r + hi * 8;
      float o = oacc[t][r] / lrun[r];
      outb[((size_t)b * S + qg) * Dm + (size_t)h * hd + t * 16 + nl] = (bf16_t)o;
    }
}

// ---------------------------------------------------------------------------
// Host orchestration
// ---------------------------------------------------------------------------
extern "C" void kernel_launch(void* const* d_in, const int* in_sizes, int n_in,
                              void* d_out, int out_size, void* d_ws, size_t ws_size,
                              hipStream_t stream) {
  const float* x           = (const float*)d_in[0];
  const unsigned char* adj = (const unsigned char*)d_in[1];
  const float* edge_attr   = (const float*)d_in[2];
  const int*   edge_idx    = (const int*)d_in[3];
  const float* n1g = (const float*)d_in[4];
  const float* n1b = (const float*)d_in[5];
  const float* Wq = (const float*)d_in[6];   const float* bq = (const float*)d_in[7];
  const float* Wk = (const float*)d_in[8];   const float* bk = (const float*)d_in[9];
  const float* Wv = (const float*)d_in[10];  const float* bv = (const float*)d_in[11];
  const float* Wo = (const float*)d_in[12];  const float* bo = (const float*)d_in[13];
  const float* n2g = (const float*)d_in[14];
  const float* n2b = (const float*)d_in[15];
  const float* W1 = (const float*)d_in[16];  const float* b1 = (const float*)d_in[17];
  const float* W2 = (const float*)d_in[18];  const float* b2 = (const float*)d_in[19];
  const float* epW1 = (const float*)d_in[20]; const float* epb1 = (const float*)d_in[21];
  const float* epW2 = (const float*)d_in[22]; const float* epb2 = (const float*)d_in[23];

  const int D  = in_sizes[4];          // 768
  const int M  = in_sizes[0] / D;      // B*S = 2048
  const int S  = in_sizes[1] / M;      // 1024
  const int Bb = M / S;                // 2
  const int E  = in_sizes[2];          // 32768
  const int H  = in_sizes[20] / 16;    // 12
  const int FF = in_sizes[17];         // 3072
  const int hd = D / H;                // 64

  // workspace partition
  char* wp = (char*)d_ws;
  auto alloc = [&](size_t bytes) -> void* {
    void* p = (void*)wp;
    wp += (bytes + 255) & ~(size_t)255;
    return p;
  };
  bf16_t* nxb    = (bf16_t*)alloc((size_t)M * D * sizeof(bf16_t));
  bf16_t* qb     = (bf16_t*)alloc((size_t)M * D * sizeof(bf16_t));
  bf16_t* kbf    = (bf16_t*)alloc((size_t)M * D * sizeof(bf16_t));
  bf16_t* vTb    = (bf16_t*)alloc((size_t)D * M * sizeof(bf16_t));  // V transposed [D, B*S]
  bf16_t* attnb  = (bf16_t*)alloc((size_t)M * D * sizeof(bf16_t));
  bf16_t* y2b    = (bf16_t*)alloc((size_t)M * D * sizeof(bf16_t));
  bf16_t* hmid   = (bf16_t*)alloc((size_t)M * FF * sizeof(bf16_t));
  bf16_t* wqT    = (bf16_t*)alloc((size_t)D * D * sizeof(bf16_t));   // [N,K]
  bf16_t* wkT    = (bf16_t*)alloc((size_t)D * D * sizeof(bf16_t));
  bf16_t* wvT    = (bf16_t*)alloc((size_t)D * D * sizeof(bf16_t));
  bf16_t* woT    = (bf16_t*)alloc((size_t)D * D * sizeof(bf16_t));
  bf16_t* w1T    = (bf16_t*)alloc((size_t)FF * D * sizeof(bf16_t));  // [FF, D]
  bf16_t* w2T    = (bf16_t*)alloc((size_t)D * FF * sizeof(bf16_t));  // [D, FF]
  float*  biasH  = (float*)alloc((size_t)H * S * S * sizeof(float));
  float*  x2     = (float*)alloc((size_t)M * D * sizeof(float));

  float* outp = (float*)d_out;

  // 1) zero the bias plane
  zero_f32_kernel<<<1024, 256, 0, stream>>>(biasH, (long)H * S * S);

  // 2) weight conversions fp32 [K,N] -> bf16 transposed [N,K]
  {
    dim3 gDD(D / 32, D / 32);
    cvt_transpose_kernel<<<gDD, 256, 0, stream>>>(Wq, wqT, D, D);
    cvt_transpose_kernel<<<gDD, 256, 0, stream>>>(Wk, wkT, D, D);
    cvt_transpose_kernel<<<gDD, 256, 0, stream>>>(Wv, wvT, D, D);
    cvt_transpose_kernel<<<gDD, 256, 0, stream>>>(Wo, woT, D, D);
    dim3 g1(FF / 32, D / 32);
    cvt_transpose_kernel<<<g1, 256, 0, stream>>>(W1, w1T, D, FF);   // [D,FF] -> [FF,D]
    dim3 g2(D / 32, FF / 32);
    cvt_transpose_kernel<<<g2, 256, 0, stream>>>(W2, w2T, FF, D);   // [FF,D] -> [D,FF]
  }

  // 3) LN1
  ln_kernel<<<M, 256, 0, stream>>>(x, n1g, n1b, nxb, D);

  // 4) QKV projections (WMMA); V written transposed for P@V fragment loads
  {
    dim3 grid(D / 128, M / 256);
    gemm_bf16_kernel<<<grid, 256, 0, stream>>>(nxb, wqT, bq, nullptr, nullptr, qb,  nullptr, M, D, D, 0);
    gemm_bf16_kernel<<<grid, 256, 0, stream>>>(nxb, wkT, bk, nullptr, nullptr, kbf, nullptr, M, D, D, 0);
    gemm_bf16_kernel<<<grid, 256, 0, stream>>>(nxb, wvT, bv, nullptr, nullptr, nullptr, vTb, M, D, D, 0);
  }

  // 5) edge bias scatter
  {
    int n = E * H;
    edge_bias_kernel<<<(n + 255) / 256, 256, 0, stream>>>(edge_attr, edge_idx,
                                                          epW1, epb1, epW2, epb2,
                                                          biasH, E, H, S);
  }

  // 6) flash attention (WMMA)
  {
    dim3 grid(S / 64, H, Bb);
    attn_kernel<<<grid, 128, 0, stream>>>(qb, kbf, vTb, adj, biasH, attnb, S, D, hd, M);
  }

  // 7) output projection + residual -> x2 (fp32)
  {
    dim3 grid(D / 128, M / 256);
    gemm_bf16_kernel<<<grid, 256, 0, stream>>>(attnb, woT, bo, x, x2, nullptr, nullptr, M, D, D, 0);
  }

  // 8) LN2
  ln_kernel<<<M, 256, 0, stream>>>(x2, n2g, n2b, y2b, D);

  // 9) MLP up + exact GELU
  {
    dim3 grid(FF / 128, M / 256);
    gemm_bf16_kernel<<<grid, 256, 0, stream>>>(y2b, w1T, b1, nullptr, nullptr, hmid, nullptr, M, FF, D, 1);
  }

  // 10) MLP down + residual -> d_out (fp32)
  {
    dim3 grid(D / 128, M / 256);
    gemm_bf16_kernel<<<grid, 256, 0, stream>>>(hmid, w2T, b2, x2, outp, nullptr, nullptr, M, D, FF, 0);
  }
}
